// QuantLlamaMLP_1812476199607
// MI455X (gfx1250) — 8-bit matrix operand order repaired and checked by reading, compiled, not run on hardware
//
#include <hip/hip_runtime.h>
#include <cstdint>
#include <cstddef>


typedef __attribute__((ext_vector_type(8))) int          v8i;
typedef __attribute__((ext_vector_type(4))) int          v4i;
typedef __attribute__((ext_vector_type(2))) int          v2i;
typedef __attribute__((ext_vector_type(4))) unsigned int v4u;

#define HAVE_TDM 0
#if defined(__has_builtin)
#if __has_builtin(__builtin_amdgcn_tensor_load_to_lds)
#undef HAVE_TDM
#define HAVE_TDM 1
#endif
#endif

__device__ __forceinline__ v8i wmma_iu8(v8i a, v8i b, v8i c) {
  return __builtin_amdgcn_wmma_i32_16x16x64_iu8(true, a, true, b, c, false, false);
}

__device__ __forceinline__ v8i load_a_frag(const int8_t* lds, int mBase, int ks, int lane) {
  const int hi = lane >> 4;
  const int m  = mBase + (lane & 15);
  const int8_t* r = lds + m * 128 + ks * 64 + hi * 8;
  v2i a0 = *(const v2i*)(r +  0);
  v2i a1 = *(const v2i*)(r + 16);
  v2i a2 = *(const v2i*)(r + 32);
  v2i a3 = *(const v2i*)(r + 48);
  v8i f;
  f[0] = a0[0]; f[1] = a0[1]; f[2] = a1[0]; f[3] = a1[1];
  f[4] = a2[0]; f[5] = a2[1]; f[6] = a3[0]; f[7] = a3[1];
  return f;
}

__device__ __forceinline__ v8i load_b_frag(const int8_t* lds, int nBase, int ks, int lane) {
  const int hi = lane >> 4;
  const int n  = nBase + (lane & 15);
  const int8_t* r = lds + n * 128 + ks * 64 + hi * 8;
  v2i b0 = *(const v2i*)(r +  0);
  v2i b1 = *(const v2i*)(r + 16);
  v2i b2 = *(const v2i*)(r + 32);
  v2i b3 = *(const v2i*)(r + 48);
  v8i f;
  f[0] = b0[0]; f[1] = b0[1]; f[2] = b1[0]; f[3] = b1[1];
  f[4] = b2[0]; f[5] = b2[1]; f[6] = b3[0]; f[7] = b3[1];
  return f;
}

#if HAVE_TDM
__device__ __forceinline__ uint32_t lds_offset(const void* p) {
  return (uint32_t)(uintptr_t)p;
}

__device__ __forceinline__ void tdm_load_tile(uint32_t ldsOff, const int8_t* gptr,
                                              uint32_t rows, uint32_t strideBytes,
                                              uint32_t rowsTotal) {
  const uint64_t ga = (uint64_t)(uintptr_t)gptr;
  v4u g0;
  g0[0] = 1u;
  g0[1] = ldsOff;
  g0[2] = (uint32_t)ga;
  g0[3] = ((uint32_t)(ga >> 32) & 0x01FFFFFFu) | (2u << 30);
  const uint32_t td0 = strideBytes;
  const uint32_t td1 = rowsTotal;
  v8i g1;
  g1[0] = 0;
  g1[1] = (int)((td0 & 0xFFFFu) << 16);
  g1[2] = (int)(((td0 >> 16) & 0xFFFFu) | ((td1 & 0xFFFFu) << 16));
  g1[3] = (int)(((td1 >> 16) & 0xFFFFu) | (128u << 16));
  g1[4] = (int)rows;
  g1[5] = (int)strideBytes;
  g1[6] = 0;
  g1[7] = 0;
  v4i z = {};
#if __clang_major__ >= 23
  v8i z8 = {};
  __builtin_amdgcn_tensor_load_to_lds(g0, g1, z, z, z8, 0);
#else
  __builtin_amdgcn_tensor_load_to_lds(g0, g1, z, z, 0);
#endif
}
#endif

__global__ void __launch_bounds__(256)
rowquant_kernel(const float* __restrict__ src, int8_t* __restrict__ dst,
                float* __restrict__ scales, int rowLen, float qmax) {
  const int row = blockIdx.x;
  const int n4  = rowLen >> 2;
  const float4* p4 = (const float4*)(src + (size_t)row * rowLen);
  float amax = 0.0f;
  for (int i = threadIdx.x; i < n4; i += 256) {
    const float4 v = p4[i];
    amax = fmaxf(amax, fmaxf(fmaxf(fabsf(v.x), fabsf(v.y)),
                             fmaxf(fabsf(v.z), fabsf(v.w))));
  }
  __shared__ float red[256];
  red[threadIdx.x] = amax;
  __syncthreads();
  for (int s = 128; s > 0; s >>= 1) {
    if ((int)threadIdx.x < s)
      red[threadIdx.x] = fmaxf(red[threadIdx.x], red[threadIdx.x + s]);
    __syncthreads();
  }
  const float scale = fmaxf(red[0] / qmax, 1e-8f);
  const float inv = 1.0f / scale;
  if (threadIdx.x == 0) scales[row] = scale;
  uint32_t* q4 = (uint32_t*)(dst + (size_t)row * rowLen);
  const float lo = -qmax - 1.0f;
  for (int i = threadIdx.x; i < n4; i += 256) {
    const float4 v = p4[i];
    const int b0 = (int)fminf(fmaxf(rintf(v.x * inv), lo), qmax);
    const int b1 = (int)fminf(fmaxf(rintf(v.y * inv), lo), qmax);
    const int b2 = (int)fminf(fmaxf(rintf(v.z * inv), lo), qmax);
    const int b3 = (int)fminf(fmaxf(rintf(v.w * inv), lo), qmax);
    q4[i] = (uint32_t)(b0 & 0xFF) | ((uint32_t)(b1 & 0xFF) << 8) |
            ((uint32_t)(b2 & 0xFF) << 16) | ((uint32_t)(b3 & 0xFF) << 24);
  }
}

__global__ void __launch_bounds__(256)
gemm_gateup_kernel(const int8_t* __restrict__ Aq, const float* __restrict__ sA,
                   const int8_t* __restrict__ Bg, const float* __restrict__ sG,
                   const int8_t* __restrict__ Bu, const float* __restrict__ sU,
                   float* __restrict__ H, int N, int K, int Mtot, int Ntot) {
  __shared__ __align__(16) int8_t lA[2][128 * 128];
  __shared__ __align__(16) int8_t lG[2][64 * 128];
  __shared__ __align__(16) int8_t lU[2][64 * 128];

  const int t     = threadIdx.x;
  const int lane  = t & 31;
  const int wave  = t >> 5;
  const int wm    = wave & 3;
  const int wn    = wave >> 2;
  const int tileM = blockIdx.y * 128;
  const int tileN = blockIdx.x * 64;
  const int nIter = K >> 7;

  v8i accg[2][2] = {};
  v8i accu[2][2] = {};

#if HAVE_TDM
  if (wave == 0) {
    tdm_load_tile(lds_offset(lA[0]), Aq + (size_t)tileM * K, 128, (uint32_t)K, (uint32_t)(Mtot - tileM));
    tdm_load_tile(lds_offset(lG[0]), Bg + (size_t)tileN * K,  64, (uint32_t)K, (uint32_t)(Ntot - tileN));
    tdm_load_tile(lds_offset(lU[0]), Bu + (size_t)tileN * K,  64, (uint32_t)K, (uint32_t)(Ntot - tileN));
  }
#endif

  for (int it = 0; it < nIter; ++it) {
    const int buf = it & 1;
#if HAVE_TDM
    if (wave == 0) {
      if (it + 1 < nIter) {
        const size_t kb2 = (size_t)(it + 1) << 7;
        tdm_load_tile(lds_offset(lA[buf ^ 1]), Aq + (size_t)tileM * K + kb2, 128, (uint32_t)K, (uint32_t)(Mtot - tileM));
        tdm_load_tile(lds_offset(lG[buf ^ 1]), Bg + (size_t)tileN * K + kb2,  64, (uint32_t)K, (uint32_t)(Ntot - tileN));
        tdm_load_tile(lds_offset(lU[buf ^ 1]), Bu + (size_t)tileN * K + kb2,  64, (uint32_t)K, (uint32_t)(Ntot - tileN));
        __builtin_amdgcn_s_wait_tensorcnt((short)3);
      } else {
        __builtin_amdgcn_s_wait_tensorcnt((short)0);
      }
    }
#else
    {
      const int kb = it << 7;
      #pragma unroll
      for (int i = 0; i < 4; ++i) {
        const int c = t + 256 * i;
        const int row = c >> 3, col = (c & 7) << 4;
        *(v4i*)(lA[buf] + row * 128 + col) = *(const v4i*)(Aq + (size_t)(tileM + row) * K + kb + col);
      }
      #pragma unroll
      for (int i = 0; i < 2; ++i) {
        const int c = t + 256 * i;
        const int row = c >> 3, col = (c & 7) << 4;
        *(v4i*)(lG[buf] + row * 128 + col) = *(const v4i*)(Bg + (size_t)(tileN + row) * K + kb + col);
        *(v4i*)(lU[buf] + row * 128 + col) = *(const v4i*)(Bu + (size_t)(tileN + row) * K + kb + col);
      }
    }
#endif
    __syncthreads();

    #pragma unroll
    for (int ks = 0; ks < 2; ++ks) {
      const v8i a0  = load_a_frag(lA[buf], wm * 32 +  0, ks, lane);
      const v8i a1  = load_a_frag(lA[buf], wm * 32 + 16, ks, lane);
      const v8i bg0 = load_b_frag(lG[buf], wn * 32 +  0, ks, lane);
      const v8i bg1 = load_b_frag(lG[buf], wn * 32 + 16, ks, lane);
      const v8i bu0 = load_b_frag(lU[buf], wn * 32 +  0, ks, lane);
      const v8i bu1 = load_b_frag(lU[buf], wn * 32 + 16, ks, lane);
      accg[0][0] = wmma_iu8(a0, bg0, accg[0][0]);
      accg[0][1] = wmma_iu8(a0, bg1, accg[0][1]);
      accg[1][0] = wmma_iu8(a1, bg0, accg[1][0]);
      accg[1][1] = wmma_iu8(a1, bg1, accg[1][1]);
      accu[0][0] = wmma_iu8(a0, bu0, accu[0][0]);
      accu[0][1] = wmma_iu8(a0, bu1, accu[0][1]);
      accu[1][0] = wmma_iu8(a1, bu0, accu[1][0]);
      accu[1][1] = wmma_iu8(a1, bu1, accu[1][1]);
    }
    __syncthreads();
  }

  const int hi = lane >> 4;
  const int nn = lane & 15;
  #pragma unroll
  for (int im = 0; im < 2; ++im) {
    float sxv[8];
    #pragma unroll
    for (int r = 0; r < 8; ++r)
      sxv[r] = sA[tileM + wm * 32 + im * 16 + hi * 8 + r];
    #pragma unroll
    for (int in = 0; in < 2; ++in) {
      const int nGlob = tileN + wn * 32 + in * 16 + nn;
      const float sg = sG[nGlob];
      const float su = sU[nGlob];
      #pragma unroll
      for (int r = 0; r < 8; ++r) {
        const int mGlob = tileM + wm * 32 + im * 16 + hi * 8 + r;
        const float g = (float)accg[im][in][r] * sxv[r] * sg;
        const float u = (float)accu[im][in][r] * sxv[r] * su;
        const float h = (g / (1.0f + __expf(-g))) * u;
        H[(size_t)mGlob * N + nGlob] = h;
      }
    }
  }
}

__global__ void __launch_bounds__(256)
gemm_down_kernel(const int8_t* __restrict__ Aq, const float* __restrict__ sA,
                 const int8_t* __restrict__ Bq, const float* __restrict__ sB,
                 float* __restrict__ Out, int N, int K, int Mtot, int Ntot) {
  __shared__ __align__(16) int8_t lA[2][128 * 128];
  __shared__ __align__(16) int8_t lB[2][64 * 128];

  const int t     = threadIdx.x;
  const int lane  = t & 31;
  const int wave  = t >> 5;
  const int wm    = wave & 3;
  const int wn    = wave >> 2;
  const int tileM = blockIdx.y * 128;
  const int tileN = blockIdx.x * 64;
  const int nIter = K >> 7;

  v8i acc[2][2] = {};

#if HAVE_TDM
  if (wave == 0) {
    tdm_load_tile(lds_offset(lA[0]), Aq + (size_t)tileM * K, 128, (uint32_t)K, (uint32_t)(Mtot - tileM));
    tdm_load_tile(lds_offset(lB[0]), Bq + (size_t)tileN * K,  64, (uint32_t)K, (uint32_t)(Ntot - tileN));
  }
#endif

  for (int it = 0; it < nIter; ++it) {
    const int buf = it & 1;
#if HAVE_TDM
    if (wave == 0) {
      if (it + 1 < nIter) {
        const size_t kb2 = (size_t)(it + 1) << 7;
        tdm_load_tile(lds_offset(lA[buf ^ 1]), Aq + (size_t)tileM * K + kb2, 128, (uint32_t)K, (uint32_t)(Mtot - tileM));
        tdm_load_tile(lds_offset(lB[buf ^ 1]), Bq + (size_t)tileN * K + kb2,  64, (uint32_t)K, (uint32_t)(Ntot - tileN));
        __builtin_amdgcn_s_wait_tensorcnt((short)2);
      } else {
        __builtin_amdgcn_s_wait_tensorcnt((short)0);
      }
    }
#else
    {
      const int kb = it << 7;
      #pragma unroll
      for (int i = 0; i < 4; ++i) {
        const int c = t + 256 * i;
        const int row = c >> 3, col = (c & 7) << 4;
        *(v4i*)(lA[buf] + row * 128 + col) = *(const v4i*)(Aq + (size_t)(tileM + row) * K + kb + col);
      }
      #pragma unroll
      for (int i = 0; i < 2; ++i) {
        const int c = t + 256 * i;
        const int row = c >> 3, col = (c & 7) << 4;
        *(v4i*)(lB[buf] + row * 128 + col) = *(const v4i*)(Bq + (size_t)(tileN + row) * K + kb + col);
      }
    }
#endif
    __syncthreads();

    #pragma unroll
    for (int ks = 0; ks < 2; ++ks) {
      const v8i a0 = load_a_frag(lA[buf], wm * 32 +  0, ks, lane);
      const v8i a1 = load_a_frag(lA[buf], wm * 32 + 16, ks, lane);
      const v8i b0 = load_b_frag(lB[buf], wn * 32 +  0, ks, lane);
      const v8i b1 = load_b_frag(lB[buf], wn * 32 + 16, ks, lane);
      acc[0][0] = wmma_iu8(a0, b0, acc[0][0]);
      acc[0][1] = wmma_iu8(a0, b1, acc[0][1]);
      acc[1][0] = wmma_iu8(a1, b0, acc[1][0]);
      acc[1][1] = wmma_iu8(a1, b1, acc[1][1]);
    }
    __syncthreads();
  }

  const int hi = lane >> 4;
  const int nn = lane & 15;
  #pragma unroll
  for (int im = 0; im < 2; ++im) {
    float sxv[8];
    #pragma unroll
    for (int r = 0; r < 8; ++r)
      sxv[r] = sA[tileM + wm * 32 + im * 16 + hi * 8 + r];
    #pragma unroll
    for (int in = 0; in < 2; ++in) {
      const int nGlob = tileN + wn * 32 + in * 16 + nn;
      const float sb = sB[nGlob];
      #pragma unroll
      for (int r = 0; r < 8; ++r) {
        const int mGlob = tileM + wm * 32 + im * 16 + hi * 8 + r;
        Out[(size_t)mGlob * N + nGlob] = (float)acc[im][in][r] * sxv[r] * sb;
      }
    }
  }
}

extern "C" void kernel_launch(void* const* d_in, const int* in_sizes, int n_in,
                              void* d_out, int out_size, void* d_ws, size_t ws_size,
                              hipStream_t stream) {
  (void)in_sizes; (void)n_in; (void)out_size; (void)ws_size;

  const int M = 2 * 2048;
  const int D = 4096;
  const int F = 11008;

  const float* x  = (const float*)d_in[0];
  const float* Wg = (const float*)d_in[1];
  const float* Wu = (const float*)d_in[2];
  const float* Wd = (const float*)d_in[3];
  float* out = (float*)d_out;

  char* p = (char*)d_ws;
  auto alloc = [&](size_t bytes) -> char* {
    char* r = p;
    p += (bytes + 255) & ~(size_t)255;
    return r;
  };
  int8_t* xq  = (int8_t*)alloc((size_t)M * D);
  float*  sx  = (float*) alloc((size_t)M * 4);
  int8_t* wgq = (int8_t*)alloc((size_t)F * D);
  float*  swg = (float*) alloc((size_t)F * 4);
  int8_t* wuq = (int8_t*)alloc((size_t)F * D);
  float*  swu = (float*) alloc((size_t)F * 4);
  int8_t* wdq = (int8_t*)alloc((size_t)D * F);
  float*  swd = (float*) alloc((size_t)D * 4);
  float*  h   = (float*) alloc((size_t)M * F * 4);
  float*  sh  = (float*) alloc((size_t)M * 4);
  int8_t* hq  = wgq;

  rowquant_kernel<<<M, 256, 0, stream>>>(x,  xq,  sx,  D, 127.0f);
  rowquant_kernel<<<F, 256, 0, stream>>>(Wg, wgq, swg, D, 7.0f);
  rowquant_kernel<<<F, 256, 0, stream>>>(Wu, wuq, swu, D, 7.0f);
  rowquant_kernel<<<D, 256, 0, stream>>>(Wd, wdq, swd, F, 7.0f);

  dim3 g1(F / 64, M / 128);
  gemm_gateup_kernel<<<g1, 256, 0, stream>>>(xq, sx, wgq, swg, wuq, swu, h, F, D, M, F);

  rowquant_kernel<<<M, 256, 0, stream>>>(h, hq, sh, F, 127.0f);

  dim3 g2(D / 64, M / 128);
  gemm_down_kernel<<<g2, 256, 0, stream>>>(hq, sh, wdq, swd, out, D, F, M, D);
}
